// RMKVBlock_22625887715955
// MI455X (gfx1250) — matrix operand order repaired and checked by reading, not compiled, not run on hardware
//
#include <hip/hip_runtime.h>

#define B_ 2
#define S_ 2048
#define M_ 64
#define D_ 1024
#define T_ (M_ + S_)   // 2112
#define H_ 16
#define HD_ 64

typedef __bf16 bf16;
typedef __attribute__((ext_vector_type(16))) __bf16 v16bf;
typedef __attribute__((ext_vector_type(8)))  float  v8f;
typedef __attribute__((ext_vector_type(4)))  unsigned v4u;

#define AS1 __attribute__((address_space(1)))
#define AS3 __attribute__((address_space(3)))

// ---------------------------------------------------------------------------
// CDNA5 async global->LDS copy (16 bytes / lane), tracked by ASYNCcnt.
// ---------------------------------------------------------------------------
static __device__ __forceinline__ void async_copy16(void* lds, const void* g) {
  asm volatile("global_load_async_to_lds_b128 %0, %1, off"
               :: "v"((AS3 void*)lds), "v"((AS1 const void*)g)
               : "memory");
}
static __device__ __forceinline__ void wait_async() {
  asm volatile("s_wait_asynccnt 0" ::: "memory");
}
static __device__ __forceinline__ void wait_ds() {
  asm volatile("s_wait_dscnt 0" ::: "memory");
}

// ---------------------------------------------------------------------------
// WMMA fragment helpers (CDNA5 wave32 layouts, ISA 7.12.2)
// ---------------------------------------------------------------------------
static __device__ __forceinline__ v16bf load_a_frag(const bf16* sm, int ld) {
  int lane = threadIdx.x & 31;
  const bf16* p = sm + (lane & 15) * ld + ((lane >> 4) << 3);
  union { v16bf v; unsigned u[8]; } f;
  const unsigned* p0 = (const unsigned*)p;
  const unsigned* p1 = (const unsigned*)(p + 16);
#pragma unroll
  for (int i = 0; i < 4; ++i) { f.u[i] = p0[i]; f.u[4 + i] = p1[i]; }
  return f.v;
}

static __device__ __forceinline__ v16bf load_b_frag(const bf16* sm, int ld) {
  int lane = threadIdx.x & 31;
  const unsigned* p = (const unsigned*)(sm + (lane & 15) * ld + ((lane >> 4) << 3));
  union { v16bf v; unsigned u[8]; } f;
#pragma unroll
  for (int i = 0; i < 4; ++i) { f.u[i] = p[i]; f.u[4 + i] = p[8 + i]; }
  return f.v;
}

static __device__ __forceinline__ v8f wmma_bf16(v16bf a, v16bf b, v8f c) {
  return __builtin_amdgcn_wmma_f32_16x16x32_bf16(false, a, false, b,
                                                 (short)0, c, false, false);
}

// ---------------------------------------------------------------------------
// bf16 GEMM: C = alpha * (A[M,K] @ B') + bias + resid
//   b_is_nk=1: B' = Bsrc^T, Bsrc laid out [N][K]   (x @ W.T form)
//   b_is_nk=0: B' = Bsrc laid out [K][N]
// A, Bsrc are bf16; C written as f32 (Cf) and/or bf16 (Cbf).
// 64x128 tile, 8 waves, K-step 32. All dims tile-aligned -> no guards.
// Tiles staged with async global->LDS copies (ASYNCcnt).
// ---------------------------------------------------------------------------
__global__ __launch_bounds__(256) void gemm_kernel(
    const bf16* __restrict__ A, int lda, long sA,
    const bf16* __restrict__ Bsrc, int ldb, long sB, int b_is_nk,
    const float* __restrict__ bias,
    const float* __restrict__ resid, int ldr,
    float* __restrict__ Cf, bf16* __restrict__ Cbf, int ldc, long sC,
    int Kd, float alpha)
{
  __shared__ __align__(16) bf16 As[64 * 40];
  __shared__ __align__(16) bf16 Bs[128 * 40];

  const int z = blockIdx.z;
  A    += (long)z * sA;
  Bsrc += (long)z * sB;

  const int x0 = blockIdx.x * 128;
  const int y0 = blockIdx.y * 64;
  const int tid = threadIdx.x;
  const int lane = tid & 31, wave = tid >> 5;

  // staging coordinates
  const int arow = tid >> 2, aseg = (tid & 3) * 8;      // A: 64 rows x 64B
  const bf16* aptr = A + (long)(y0 + arow) * lda + aseg;
  bf16* alds = &As[arow * 40 + aseg];
  const int brow = tid >> 1, bseg = (tid & 1) * 16;     // B[n][k]: 128 x 64B
  const bf16* bptr = Bsrc + (long)(x0 + brow) * ldb + bseg;
  bf16* blds = &Bs[brow * 40 + bseg];
  const int tk = tid >> 3, tn0 = (tid & 7) * 16;        // B[k][n] transpose path
  const bf16* tptr = Bsrc + (long)tk * ldb + x0 + tn0;

  v8f acc[4] = {};

  for (int k0 = 0; k0 < Kd; k0 += 32) {
    async_copy16(alds, aptr + k0);                      // A tile: 1 op/thread
    if (b_is_nk) {                                      // B tile: 2 ops/thread
      async_copy16(blds,     bptr + k0);
      async_copy16(blds + 8, bptr + k0 + 8);
    } else {                                            // transpose [K][N]
      const bf16* tp = tptr + (long)k0 * ldb;
      union { v4u q; bf16 h[8]; } u0, u1;
      u0.q = *(const v4u*)tp;
      u1.q = *(const v4u*)(tp + 8);
#pragma unroll
      for (int e = 0; e < 8; ++e) {
        Bs[(tn0 + e)     * 40 + tk] = u0.h[e];
        Bs[(tn0 + 8 + e) * 40 + tk] = u1.h[e];
      }
    }
    wait_async();
    __syncthreads();

    v16bf bfg = load_b_frag(Bs + wave * 16 * 40, 40);
    v16bf af[4];
#pragma unroll
    for (int r = 0; r < 4; ++r) af[r] = load_a_frag(As + r * 16 * 40, 40);
#pragma unroll
    for (int r = 0; r < 4; ++r) acc[r] = wmma_bf16(af[r], bfg, acc[r]);
    __syncthreads();
  }

  const int col = x0 + wave * 16 + (lane & 15);
  const int hi  = lane >> 4;
  float badd = bias ? bias[col] : 0.f;
#pragma unroll
  for (int r = 0; r < 4; ++r) {
#pragma unroll
    for (int rr = 0; rr < 8; ++rr) {
      int m = y0 + r * 16 + hi * 8 + rr;
      float v = acc[r][rr] * alpha + badd;
      if (resid) v += resid[(long)m * ldr + col];
      if (Cf)  Cf [(long)z * sC + (long)m * ldc + col] = v;
      if (Cbf) Cbf[(long)z * sC + (long)m * ldc + col] = (bf16)v;
    }
  }
}

// ---------------------------------------------------------------------------
// Flash attention on bf16 q/k/v: grid (T/64, H, B), 4 waves x 16 q-rows,
// kv chunks of 32. Q/K tiles staged via async copies; V transposed manually.
// Output ctx written as bf16 (sole consumer is the bf16 out-proj GEMM).
// ---------------------------------------------------------------------------
__global__ __launch_bounds__(128) void attn_kernel(
    const bf16* __restrict__ Q, const bf16* __restrict__ K,
    const bf16* __restrict__ V, bf16* __restrict__ O)
{
  __shared__ __align__(16) bf16 Qs[64 * 72];
  __shared__ __align__(16) bf16 Ks[32 * 72];
  __shared__ __align__(16) bf16 Vt[64 * 40];      // transposed: [d][j]
  __shared__ __align__(16) bf16 Ps[4 * 16 * 40];  // per-wave P staging

  const int b = blockIdx.z, h = blockIdx.y;
  const int q0 = blockIdx.x * 64;
  const int tid = threadIdx.x, lane = tid & 31, wave = tid >> 5;
  const long headoff = (long)h * HD_;
  const float scale = 0.125f;   // hd^-0.5, hd=64

  {   // 64x64 Q tile: 4 async 16B ops per thread
    int row = tid >> 1, off = (tid & 1) * 32;
    const bf16* qp = Q + ((long)(b * T_ + q0 + row)) * D_ + headoff + off;
    bf16* ql = &Qs[row * 72 + off];
#pragma unroll
    for (int g = 0; g < 4; ++g) async_copy16(ql + g * 8, qp + g * 8);
  }
  wait_async();
  __syncthreads();

  v16bf aq0 = load_a_frag(Qs + wave * 16 * 72, 72);        // d 0..31
  v16bf aq1 = load_a_frag(Qs + wave * 16 * 72 + 32, 72);   // d 32..63

  v8f Oacc[4] = {};
  float mrun[8], lrun[8];
#pragma unroll
  for (int r = 0; r < 8; ++r) { mrun[r] = -1e30f; lrun[r] = 0.f; }

  const int qmax = q0 + 63;
  const int nkv = (q0 < M_) ? T_ : ((qmax + 1 < T_) ? (qmax + 1) : T_);
  bf16* Pst = Ps + wave * 16 * 40;

  const int krow = tid >> 2, kseg = (tid & 3) * 8;   // K chunk: 32 rows x 128B
  const int vj = tid >> 2, vd0 = (tid & 3) * 16;     // V transpose coords

  for (int c0 = 0; c0 < nkv; c0 += 32) {
    {   // K chunk [j][d] 32x64: 2 async ops per thread
      const bf16* kp = K + ((long)(b * T_ + c0 + krow)) * D_ + headoff + kseg;
      bf16* kl = &Ks[krow * 72 + kseg];
      async_copy16(kl, kp);
      async_copy16(kl + 32, kp + 32);
    }
    {   // V chunk -> Vt[d][j] transposed (b128 loads + b16 scatter)
      const bf16* vp = V + ((long)(b * T_ + c0 + vj)) * D_ + headoff + vd0;
      union { v4u q; bf16 h[8]; } u0, u1;
      u0.q = *(const v4u*)vp;
      u1.q = *(const v4u*)(vp + 8);
#pragma unroll
      for (int e = 0; e < 8; ++e) {
        Vt[(vd0 + e)     * 40 + vj] = u0.h[e];
        Vt[(vd0 + 8 + e) * 40 + vj] = u1.h[e];
      }
    }
    wait_async();
    __syncthreads();

    // S = Q K^T : 16x32 score tile per wave (K-dim 64 -> 2 wmma per half)
    v16bf bk00 = load_b_frag(Ks, 72);
    v16bf bk01 = load_b_frag(Ks + 32, 72);
    v16bf bk10 = load_b_frag(Ks + 16 * 72, 72);
    v16bf bk11 = load_b_frag(Ks + 16 * 72 + 32, 72);
    v8f s0{}, s1{};
    s0 = wmma_bf16(aq0, bk00, s0);
    s0 = wmma_bf16(aq1, bk01, s0);
    s1 = wmma_bf16(aq0, bk10, s1);
    s1 = wmma_bf16(aq1, bk11, s1);

    const int hi = lane >> 4, colj = lane & 15;
#pragma unroll
    for (int r = 0; r < 8; ++r) {
      int qi = q0 + wave * 16 + r + 8 * hi;
      int j0 = c0 + colj;
      int j1 = c0 + 16 + colj;
      float sc0 = s0[r] * scale;
      float sc1 = s1[r] * scale;
      bool mk0 = (qi >= M_) && (j0 >= M_) && (j0 > qi);
      bool mk1 = (qi >= M_) && (j1 >= M_) && (j1 > qi);
      if (mk0) sc0 = -1e30f;
      if (mk1) sc1 = -1e30f;
      float mx = fmaxf(sc0, sc1);
#pragma unroll
      for (int d = 1; d < 16; d <<= 1) mx = fmaxf(mx, __shfl_xor(mx, d, 32));
      float mnew = fmaxf(mrun[r], mx);
      float corr = __expf(mrun[r] - mnew);
      float p0 = mk0 ? 0.f : __expf(sc0 - mnew);
      float p1 = mk1 ? 0.f : __expf(sc1 - mnew);
      float ps = p0 + p1;
#pragma unroll
      for (int d = 1; d < 16; d <<= 1) ps += __shfl_xor(ps, d, 32);
      lrun[r] = lrun[r] * corr + ps;
      mrun[r] = mnew;
#pragma unroll
      for (int ct = 0; ct < 4; ++ct) Oacc[ct][r] *= corr;
      Pst[(r + 8 * hi) * 40 + colj]      = (bf16)p0;
      Pst[(r + 8 * hi) * 40 + 16 + colj] = (bf16)p1;
    }
    wait_ds();                                     // LDS P staging visible

    v16bf pa = load_a_frag(Pst, 40);               // P as A-frag, K(=j)=0..31
#pragma unroll
    for (int ct = 0; ct < 4; ++ct) {
      v16bf bv = load_b_frag(Vt + ct * 16 * 40, 40);
      Oacc[ct] = wmma_bf16(pa, bv, Oacc[ct]);
    }
    __syncthreads();
  }

  const int hi = lane >> 4, coln = lane & 15;
#pragma unroll
  for (int r = 0; r < 8; ++r) {
    int qi = q0 + wave * 16 + r + 8 * hi;
    float inv = 1.f / lrun[r];
#pragma unroll
    for (int ct = 0; ct < 4; ++ct)
      O[((long)(b * T_ + qi)) * D_ + headoff + ct * 16 + coln] =
          (bf16)(Oacc[ct][r] * inv);
  }
}

// ---------------------------------------------------------------------------
// Elementwise helpers
// ---------------------------------------------------------------------------
__global__ void cvt_bf16_kernel(const float* __restrict__ in,
                                bf16* __restrict__ out, long n) {
  long i = blockIdx.x * (long)blockDim.x + threadIdx.x;
  if (i < n) out[i] = (bf16)in[i];
}

__global__ void concat_x_kernel(const float* __restrict__ tokens,
                                const float* __restrict__ memory,
                                float* __restrict__ x, bf16* __restrict__ xbf,
                                long n) {
  long i = blockIdx.x * (long)blockDim.x + threadIdx.x;
  if (i >= n) return;
  int d = (int)(i % D_);
  long td = i / D_;
  int t = (int)(td % T_);
  int b = (int)(td / T_);
  float v = (t < M_) ? memory[((long)(b * M_ + t)) * D_ + d]
                     : tokens[((long)(b * S_ + (t - M_))) * D_ + d];
  x[i] = v;
  xbf[i] = (bf16)v;
}

__global__ void split_tok_kernel(const float* __restrict__ xout,
                                 float* __restrict__ tok_out,
                                 bf16* __restrict__ tok_bf, long n) {
  long i = blockIdx.x * (long)blockDim.x + threadIdx.x;
  if (i >= n) return;
  int d = (int)(i % D_);
  long sd = i / D_;
  int s = (int)(sd % S_);
  int b = (int)(sd / S_);
  float v = xout[((long)(b * T_ + M_ + s)) * D_ + d];
  tok_out[i] = v;
  tok_bf[i] = (bf16)v;
}

__global__ __launch_bounds__(256) void softmax_rows_kernel(
    float* __restrict__ p, bf16* __restrict__ pbf, int cols) {
  __shared__ float red[256];
  long base = (long)blockIdx.x * cols;
  float* row = p + base;
  int tid = threadIdx.x;
  float mx = -1e30f;
  for (int c = tid; c < cols; c += 256) mx = fmaxf(mx, row[c]);
  red[tid] = mx; __syncthreads();
  for (int s = 128; s > 0; s >>= 1) {
    if (tid < s) red[tid] = fmaxf(red[tid], red[tid + s]);
    __syncthreads();
  }
  mx = red[0]; __syncthreads();
  float sum = 0.f;
  for (int c = tid; c < cols; c += 256) {
    float e = __expf(row[c] - mx);
    row[c] = e;
    sum += e;
  }
  red[tid] = sum; __syncthreads();
  for (int s = 128; s > 0; s >>= 1) {
    if (tid < s) red[tid] += red[tid + s];
    __syncthreads();
  }
  float inv = 1.f / red[0];
  for (int c = tid; c < cols; c += 256) {
    float v = row[c] * inv;
    row[c] = v;
    pbf[base + c] = (bf16)v;
  }
}

__global__ void combined_kernel(const float* __restrict__ xout,
                                const float* __restrict__ summary,
                                bf16* __restrict__ comb, long n) {
  long i = blockIdx.x * (long)blockDim.x + threadIdx.x;
  if (i >= n) return;
  int c = (int)(i % (2 * D_));
  long mrow = i / (2 * D_);
  int m = (int)(mrow % M_);
  int b = (int)(mrow / M_);
  float v = (c < D_) ? xout[((long)(b * T_ + m)) * D_ + c]
                     : summary[((long)(b * M_ + m)) * D_ + (c - D_)];
  comb[i] = (bf16)v;
}

__global__ void gate_kernel(const float* __restrict__ update,
                            const float* __restrict__ xout,
                            float* __restrict__ newmem, long n) {
  long i = blockIdx.x * (long)blockDim.x + threadIdx.x;
  if (i >= n) return;
  int d = (int)(i % D_);
  long md = i / D_;
  int m = (int)(md % M_);
  int b = (int)(md / M_);
  const float* up = update + ((long)(b * M_ + m)) * (2 * D_);
  float g   = 1.f / (1.f + __expf(-up[d]));
  float cd  = up[D_ + d];
  float sil = cd / (1.f + __expf(-cd));
  float mo  = xout[((long)(b * T_ + m)) * D_ + d];
  newmem[i] = mo * g + (1.f - g) * sil;
}

// ---------------------------------------------------------------------------
static inline unsigned cdiv256(long n) { return (unsigned)((n + 255) / 256); }

extern "C" void kernel_launch(void* const* d_in, const int* in_sizes, int n_in,
                              void* d_out, int out_size, void* d_ws, size_t ws_size,
                              hipStream_t stream) {
  (void)in_sizes; (void)n_in; (void)out_size; (void)ws_size;
  const float* tokens = (const float*)d_in[0];
  const float* memory = (const float*)d_in[1];
  const float* Wq = (const float*)d_in[2];
  const float* bq = (const float*)d_in[3];
  const float* Wk = (const float*)d_in[4];
  const float* bk = (const float*)d_in[5];
  const float* Wv = (const float*)d_in[6];
  const float* bv = (const float*)d_in[7];
  const float* Wo = (const float*)d_in[8];
  const float* bo = (const float*)d_in[9];
  const float* pool_q = (const float*)d_in[10];
  const float* cell_W = (const float*)d_in[11];
  const float* cell_b = (const float*)d_in[12];

  float* out = (float*)d_out;
  float* tok_out = out;                             // B*S*D f32
  float* new_mem = out + (long)B_ * S_ * D_;        // B*M*D f32

  const long nBTD = (long)B_ * T_ * D_;             // 4,325,376
  const long nDD  = (long)D_ * D_;
  float* ws = (float*)d_ws;
  float* x      = ws; ws += nBTD;
  float* xout   = ws; ws += nBTD;
  float* psc    = ws; ws += (long)B_ * M_ * S_;
  float* summ   = ws; ws += (long)B_ * M_ * D_;
  float* upd    = ws; ws += (long)B_ * M_ * 2 * D_;
  bf16* x_bf    = (bf16*)ws; ws += nBTD / 2;
  bf16* q_bf    = (bf16*)ws; ws += nBTD / 2;
  bf16* k_bf    = (bf16*)ws; ws += nBTD / 2;
  bf16* v_bf    = (bf16*)ws; ws += nBTD / 2;
  bf16* ctx_bf  = (bf16*)ws; ws += nBTD / 2;
  bf16* tok_bf  = (bf16*)ws; ws += (long)B_ * S_ * D_ / 2;
  bf16* wq_bf   = (bf16*)ws; ws += nDD / 2;
  bf16* wk_bf   = (bf16*)ws; ws += nDD / 2;
  bf16* wv_bf   = (bf16*)ws; ws += nDD / 2;
  bf16* wo_bf   = (bf16*)ws; ws += nDD / 2;
  bf16* cw_bf   = (bf16*)ws; ws += (long)4 * D_ * D_ / 2;
  bf16* pq_bf   = (bf16*)ws; ws += (long)M_ * D_ / 2;
  bf16* psc_bf  = (bf16*)ws; ws += (long)B_ * M_ * S_ / 2;
  bf16* comb_bf = (bf16*)ws; ws += (long)B_ * M_ * 2 * D_ / 2;

  // 1. x = concat(memory, tokens)  (f32 + bf16)
  concat_x_kernel<<<cdiv256(nBTD), 256, 0, stream>>>(tokens, memory, x, x_bf, nBTD);

  // 2. one-time weight conversions to bf16
  cvt_bf16_kernel<<<cdiv256(nDD), 256, 0, stream>>>(Wq, wq_bf, nDD);
  cvt_bf16_kernel<<<cdiv256(nDD), 256, 0, stream>>>(Wk, wk_bf, nDD);
  cvt_bf16_kernel<<<cdiv256(nDD), 256, 0, stream>>>(Wv, wv_bf, nDD);
  cvt_bf16_kernel<<<cdiv256(nDD), 256, 0, stream>>>(Wo, wo_bf, nDD);
  cvt_bf16_kernel<<<cdiv256(4 * nDD), 256, 0, stream>>>(cell_W, cw_bf, 4 * nDD);
  cvt_bf16_kernel<<<cdiv256((long)M_ * D_), 256, 0, stream>>>(pool_q, pq_bf,
                                                              (long)M_ * D_);

  // 3. Q/K/V projections -> bf16 outputs (attention is the only consumer)
  dim3 gproj(D_ / 128, (B_ * T_) / 64, 1);
  gemm_kernel<<<gproj, 256, 0, stream>>>(x_bf, D_, 0, wq_bf, D_, 0, 1, bq,
      nullptr, 0, nullptr, q_bf, D_, 0, D_, 1.f);
  gemm_kernel<<<gproj, 256, 0, stream>>>(x_bf, D_, 0, wk_bf, D_, 0, 1, bk,
      nullptr, 0, nullptr, k_bf, D_, 0, D_, 1.f);
  gemm_kernel<<<gproj, 256, 0, stream>>>(x_bf, D_, 0, wv_bf, D_, 0, 1, bv,
      nullptr, 0, nullptr, v_bf, D_, 0, D_, 1.f);

  // 4. attention (bf16 in, bf16 ctx out)
  attn_kernel<<<dim3(T_ / 64, H_, B_), 128, 0, stream>>>(q_bf, k_bf, v_bf, ctx_bf);

  // 5. out projection + residual(x) -> xout f32
  gemm_kernel<<<gproj, 256, 0, stream>>>(ctx_bf, D_, 0, wo_bf, D_, 0, 1, bo,
      x, D_, xout, nullptr, D_, 0, D_, 1.f);

  // 6. tok_out = xout[:, M:, :]  (f32 to d_out + bf16 for pooling GEMMs)
  split_tok_kernel<<<cdiv256((long)B_ * S_ * D_), 256, 0, stream>>>(
      xout, tok_out, tok_bf, (long)B_ * S_ * D_);

  // 7. pool scores: pool_q @ tok_out^T * D^-0.5  (per batch)
  gemm_kernel<<<dim3(S_ / 128, 1, B_), 256, 0, stream>>>(
      pq_bf, D_, 0, tok_bf, D_, (long)S_ * D_, 1, nullptr,
      nullptr, 0, psc, nullptr, S_, (long)M_ * S_, D_, 0.03125f);

  // 8. softmax over S (also emits bf16 weights)
  softmax_rows_kernel<<<B_ * M_, 256, 0, stream>>>(psc, psc_bf, S_);

  // 9. summary = pweights @ tok_out   ([K][N] path)
  gemm_kernel<<<dim3(D_ / 128, 1, B_), 256, 0, stream>>>(
      psc_bf, S_, (long)M_ * S_, tok_bf, D_, (long)S_ * D_, 0, nullptr,
      nullptr, 0, summ, nullptr, D_, (long)M_ * D_, S_, 1.f);

  // 10. combined = [mem_out | summary] -> bf16
  combined_kernel<<<cdiv256((long)B_ * M_ * 2 * D_), 256, 0, stream>>>(
      xout, summ, comb_bf, (long)B_ * M_ * 2 * D_);

  // 11. update = combined @ cell_W^T + cell_b
  gemm_kernel<<<dim3((2 * D_) / 128, (B_ * M_) / 64, 1), 256, 0, stream>>>(
      comb_bf, 2 * D_, 0, cw_bf, 2 * D_, 0, 1, cell_b,
      nullptr, 0, upd, nullptr, 2 * D_, 0, 2 * D_, 1.f);

  // 12. gating -> new_memory
  gate_kernel<<<cdiv256((long)B_ * M_ * D_), 256, 0, stream>>>(
      upd, xout, new_mem, (long)B_ * M_ * D_);
}
